// SelfAttention_27882927685990
// MI455X (gfx1250) — matrix operand order repaired and checked by reading, not compiled, not run on hardware
//
#include <hip/hip_runtime.h>
#include <hip/hip_bf16.h>

// ---------------------------------------------------------------------------
// Self-attention for MI455X (gfx1250, wave32, WMMA f16 16x16x32 + TDM).
//   out[0 : N*N]        = softmax(Q K^T / sqrt(D))      (fp32)
//   out[N*N : N*N+N*D]  = softmax(...) @ V              (fp32)
// Workspace: Qh (f16 NxD) | Kh (f16 NxD) | Vt (f16 DxN, transposed) = 24 MB
// ---------------------------------------------------------------------------

#define AN 8192   // N
#define AD 512    // D

typedef __attribute__((ext_vector_type(16))) _Float16 v16h;
typedef __attribute__((ext_vector_type(8)))  _Float16 v8h;
typedef __attribute__((ext_vector_type(8)))  float    v8f;
typedef __attribute__((ext_vector_type(4)))  unsigned int v4u;
typedef __attribute__((ext_vector_type(8)))  int      v8i;
typedef __attribute__((ext_vector_type(4)))  int      v4i;

union AFrag { v16h v; v8h h[2]; };
union BFrag { v16h v; v8h h[2]; };

// ---------------------------------------------------------------------------
// TDM: load a 32(k) x 64(row) f16 tile of a [8192 x 512] row-major tensor
// into LDS, padding 16B after every 64B row => 80B LDS row stride (bank-
// conflict-free for the 16-lane B-fragment reads; 80 = 16*5, 5 coprime 16).
// D# per CDNA5 ISA ch. 8: group0 = {count/lds_addr/global_addr/type},
// group1 = {flags+pad | dims | tile dims | stride}.
// ---------------------------------------------------------------------------
__device__ __forceinline__ void tdm_load_k_tile(unsigned lds_addr,
                                                const _Float16* gptr) {
  unsigned long long ga = (unsigned long long)(uintptr_t)gptr;
  v4u g0;
  g0.x = 1u;                                            // count=1 (valid)
  g0.y = lds_addr;                                      // LDS byte address
  g0.z = (unsigned)(ga & 0xFFFFFFFFu);                  // global_addr[31:0]
  g0.w = (unsigned)((ga >> 32) & 0x1FFFFFFu) | (2u << 30); // addr[56:32]|type=2
  v8i g1;
  g1[0] = (1 << 16)        // data_size = 1 (2 bytes, f16)
        | (1 << 20)        // pad_enable
        | (3 << 22)        // pad_interval: 16 DWORDs (64B) between pads
        | (3 << 25);       // pad_amount: 4 DWORDs (16B)
  g1[1] = (int)(512u << 16);   // tensor_dim0 = 512   (bits 79:48, low half)
  g1[2] = (int)(8192u << 16);  // tensor_dim1 = 8192  (bits 111:80, low half)
  g1[3] = (int)(32u << 16);    // tile_dim0 = 32      (bits 127:112)
  g1[4] = 64;                  // tile_dim1 = 64, tile_dim2 = 0
  g1[5] = 512;                 // tensor_dim0_stride = 512
  g1[6] = 0;                   // stride hi / tensor_dim1_stride lo
  g1[7] = 0;                   // tensor_dim1_stride hi
  v4i z = {0, 0, 0, 0};
#if defined(__clang_major__) && (__clang_major__ >= 23)
  v8i z8 = {0, 0, 0, 0, 0, 0, 0, 0};
  __builtin_amdgcn_tensor_load_to_lds(g0, g1, z, z, z8, 0);
#else
  __builtin_amdgcn_tensor_load_to_lds(g0, g1, z, z, 0);
#endif
}

// ---------------------------------------------------------------------------
// Kernel 1: convert Q,K to f16 row-major; V to f16 TRANSPOSED (Vt[d][n]).
// ---------------------------------------------------------------------------
__global__ __launch_bounds__(256) void attn_cvt_kernel(
    const float* __restrict__ q, const float* __restrict__ k,
    const float* __restrict__ v,
    _Float16* __restrict__ qh, _Float16* __restrict__ kh,
    _Float16* __restrict__ vt) {
  const int i = blockIdx.x * 256 + threadIdx.x;   // i in [0, N*D)
  qh[i] = (_Float16)q[i];
  kh[i] = (_Float16)k[i];
  const int r = i >> 9;         // row in V  (N dim)
  const int c = i & (AD - 1);   // col in V  (D dim)
  vt[(size_t)c * AN + r] = (_Float16)v[i];
}

// ---------------------------------------------------------------------------
// Kernel 2: S = (Q K^T) / sqrt(D).  Wave tile 16x64, WG = 8 waves = 16x512.
// B operand (K rows) streamed by the Tensor Data Mover into per-wave
// double-buffered LDS (80KB/WG); no cross-wave sharing => no barriers,
// synchronization purely via TENSORcnt / DScnt.
// Grid: (N/16, N/512), dynamic LDS = 81920 bytes.
// ---------------------------------------------------------------------------
__global__ __launch_bounds__(256) void attn_qk_kernel(
    const _Float16* __restrict__ Qh, const _Float16* __restrict__ Kh,
    float* __restrict__ S) {
  extern __shared__ _Float16 sh[];   // 8 waves * 2 bufs * 64 rows * 80B
  const int lane = threadIdx.x & 31;
  const int wave = threadIdx.x >> 5;
  const int m0   = blockIdx.x * 16;
  const int n0   = blockIdx.y * 512 + wave * 64;
  const int mr   = lane & 15;
  const int half = lane >> 4;

  v8f acc[4] = {};

  const _Float16* qrow = Qh + (size_t)(m0 + mr) * AD + half * 8;
  const _Float16* gb   = Kh + (size_t)n0 * AD;          // wave's 64 K-rows
  const unsigned  lds0 = (unsigned)(uintptr_t)sh + (unsigned)wave * 10240u;
  _Float16* lw = sh + (size_t)wave * 5120;              // element offset

  // Prime the double buffer: tiles for k-steps 0 and 1.
  tdm_load_k_tile(lds0,         gb);
  tdm_load_k_tile(lds0 + 5120u, gb + 32);

#pragma unroll
  for (int ki = 0; ki < 16; ++ki) {
    const int k0 = ki * 32;
    AFrag a;
    a.h[0] = *(const v8h*)(qrow + k0);
    a.h[1] = *(const v8h*)(qrow + k0 + 16);

    if (ki < 14) __builtin_amdgcn_s_wait_tensorcnt(1);  // tile ki landed
    else         __builtin_amdgcn_s_wait_tensorcnt(0);  // drain tail
    __asm__ volatile("" ::: "memory");

    const _Float16* lbuf = lw + (ki & 1) * 2560;        // 5120B buffer
#pragma unroll
    for (int t = 0; t < 4; ++t) {
      // B column n = padded LDS row (t*16+mr); K = half*8 + [0..7], 16 + half*8 + [0..7]
      const _Float16* row = lbuf + (t * 16 + mr) * 40 + half * 8;
      BFrag b;
      b.h[0] = *(const v8h*)(row);
      b.h[1] = *(const v8h*)(row + 16);
      acc[t] = __builtin_amdgcn_wmma_f32_16x16x32_f16(
          false, a.v, false, b.v, (short)0, acc[t], false, false);
    }

    if (ki + 2 < 16) {
      // LDS reads of buffer (ki&1) must drain before TDM overwrites it.
      __asm__ volatile("s_wait_dscnt 0" ::: "memory");
      tdm_load_k_tile(lds0 + (unsigned)(ki & 1) * 5120u, gb + (ki + 2) * 32);
    }
  }

  const float scale = 0.044194173824159216f;  // 1/sqrt(512)
#pragma unroll
  for (int t = 0; t < 4; ++t) {
    // C/D layout: M = m0 + 8*half + r, Ncol = n0 + t*16 + (lane&15)
    float* out = S + (size_t)(m0 + 8 * half) * AN + n0 + t * 16 + mr;
#pragma unroll
    for (int r = 0; r < 8; ++r) out[(size_t)r * AN] = acc[t][r] * scale;
  }
}

// ---------------------------------------------------------------------------
// Kernel 3: in-place row softmax. One block (1024 thr) per row; the whole
// 8192-float row lives in registers (8/thread): 1 read + 1 write of 256 MB.
// ---------------------------------------------------------------------------
__global__ __launch_bounds__(1024) void attn_softmax_kernel(float* __restrict__ S) {
  float* p = S + (size_t)blockIdx.x * AN;
  const int t    = threadIdx.x;
  const int lane = t & 31;
  const int wave = t >> 5;
  __shared__ float red[32];

  float vals[8];
  float m = -3.402823466e38f;
#pragma unroll
  for (int i = 0; i < 8; ++i) {
    vals[i] = p[t * 8 + i];
    m = fmaxf(m, vals[i]);
  }
#pragma unroll
  for (int off = 16; off; off >>= 1) m = fmaxf(m, __shfl_xor(m, off, 32));
  if (lane == 0) red[wave] = m;
  __syncthreads();
  if (wave == 0) {
    float x = red[lane];
#pragma unroll
    for (int off = 16; off; off >>= 1) x = fmaxf(x, __shfl_xor(x, off, 32));
    if (lane == 0) red[0] = x;
  }
  __syncthreads();
  const float rowmax = red[0];
  __syncthreads();

  float s = 0.f;
#pragma unroll
  for (int i = 0; i < 8; ++i) {
    vals[i] = __expf(vals[i] - rowmax);
    s += vals[i];
  }
#pragma unroll
  for (int off = 16; off; off >>= 1) s += __shfl_xor(s, off, 32);
  if (lane == 0) red[wave] = s;
  __syncthreads();
  if (wave == 0) {
    float x = red[lane];
#pragma unroll
    for (int off = 16; off; off >>= 1) x += __shfl_xor(x, off, 32);
    if (lane == 0) red[0] = x;
  }
  __syncthreads();
  const float inv = 1.0f / red[0];

#pragma unroll
  for (int i = 0; i < 8; ++i) p[t * 8 + i] = vals[i] * inv;
}

// ---------------------------------------------------------------------------
// Kernel 4: O = P @ V  using Vt (f16, [D][N]).  Wave tile 16x64 over D,
// WG = 8 waves covers the full D=512 for 16 rows. Grid: N/16.
// P (fp32, 256MB stream) converted to f16 in-register per k-step.
// ---------------------------------------------------------------------------
__global__ __launch_bounds__(256) void attn_pv_kernel(
    const float* __restrict__ P, const _Float16* __restrict__ Vt,
    float* __restrict__ O) {
  const int lane = threadIdx.x & 31;
  const int wave = threadIdx.x >> 5;
  const int m0   = blockIdx.x * 16;
  const int n0   = wave * 64;
  const int mr   = lane & 15;
  const int half = lane >> 4;

  v8f acc[4] = {};
  const float* prow = P + (size_t)(m0 + mr) * AN + half * 8;

  for (int k0 = 0; k0 < AN; k0 += 32) {
    v8f x0 = *(const v8f*)(prow + k0);
    v8f x1 = *(const v8f*)(prow + k0 + 16);
    if (k0 + 32 < AN) __builtin_prefetch(prow + k0 + 32, 0, 1);
    AFrag a;
#pragma unroll
    for (int i = 0; i < 8; ++i) {
      a.v[i]     = (_Float16)x0[i];
      a.v[8 + i] = (_Float16)x1[i];
    }
#pragma unroll
    for (int t = 0; t < 4; ++t) {
      const _Float16* vrow =
          Vt + (size_t)(n0 + t * 16 + mr) * AN + k0 + half * 8;
      BFrag b;
      b.h[0] = *(const v8h*)vrow;
      b.h[1] = *(const v8h*)(vrow + 16);
      acc[t] = __builtin_amdgcn_wmma_f32_16x16x32_f16(
          false, a.v, false, b.v, (short)0, acc[t], false, false);
    }
  }

#pragma unroll
  for (int t = 0; t < 4; ++t) {
    float* out = O + (size_t)(m0 + 8 * half) * AD + n0 + t * 16 + mr;
#pragma unroll
    for (int r = 0; r < 8; ++r) out[(size_t)r * AD] = acc[t][r];
  }
}

// ---------------------------------------------------------------------------
extern "C" void kernel_launch(void* const* d_in, const int* in_sizes, int n_in,
                              void* d_out, int out_size, void* d_ws,
                              size_t ws_size, hipStream_t stream) {
  const float* q = (const float*)d_in[0];
  const float* k = (const float*)d_in[1];
  const float* v = (const float*)d_in[2];
  float* out = (float*)d_out;

  _Float16* qh = (_Float16*)d_ws;
  _Float16* kh = qh + (size_t)AN * AD;
  _Float16* vt = kh + (size_t)AN * AD;   // 3 * 8 MB = 24 MB workspace

  float* Smat = out;                       // [N, N] attention weights
  float* Omat = out + (size_t)AN * AN;     // [N, D] attention embedding

  // 1) fp32 -> f16 conversion (+ V transpose)
  attn_cvt_kernel<<<(AN * AD) / 256, 256, 0, stream>>>(q, k, v, qh, kh, vt);

  // 2) S = Q K^T / sqrt(D)  (TDM-staged B operand, 80KB dynamic LDS)
  attn_qk_kernel<<<dim3(AN / 16, AN / 512), 256, 81920, stream>>>(qh, kh, Smat);

  // 3) row softmax in place
  attn_softmax_kernel<<<AN, 1024, 0, stream>>>(Smat);

  // 4) O = P V
  attn_pv_kernel<<<AN / 16, 256, 0, stream>>>(Smat, vt, Omat);
}
